// Glm4MoeDecoderLayer_63221918597201
// MI455X (gfx1250) — matrix operand order repaired and checked by reading, not compiled, not run on hardware
//
#include <hip/hip_runtime.h>
#include <hip/hip_bf16.h>

typedef __attribute__((ext_vector_type(16))) _Float16 v16h;
typedef __attribute__((ext_vector_type(8)))  _Float16 v8h;
typedef __attribute__((ext_vector_type(8)))  float    v8f;

#define LDS_FENCE() asm volatile("s_wait_dscnt 0" ::: "memory")

static __device__ __forceinline__ v8f wmma32(v16h a, v16h b, v8f c) {
  return __builtin_amdgcn_wmma_f32_16x16x32_f16(false, a, false, b, (short)0, c, false, false);
}

static __device__ __forceinline__ v16h cat16(v8h x0, v8h x1) {
  v16h r;
#pragma unroll
  for (int e = 0; e < 8; ++e) { r[e] = x0[e]; r[e + 8] = x1[e]; }
  return r;
}

// A fragment (16x32 f16) from row-major f16: two contiguous 16B loads per lane.
// element e -> k = k0 + e + 16*hi
static __device__ __forceinline__ v16h frag_a16(const _Float16* arow, int k0, int hi) {
  const _Float16* p = arow + k0 + (hi << 4);
  return cat16(*(const v8h*)p, *(const v8h*)(p + 8));
}

// B fragment (32x16 f16) from TRANSPOSED f16 Bt[N][K]: two contiguous 16B loads per lane.
// element e -> k = k0 + e + 16*hi, col = lane%16
static __device__ __forceinline__ v16h frag_bt(const _Float16* Bt, int ldk, int k0, int col, int hi) {
  const _Float16* p = Bt + (size_t)col * ldk + k0 + (hi << 4);
  return cat16(*(const v8h*)p, *(const v8h*)(p + 8));
}

// ---------------- transpose + f32->f16 convert: dst[N][K] = (f16)src[K][N] ----------------
__global__ __launch_bounds__(256) void transpose_cvt_kernel(const float* __restrict__ src,
                                                            _Float16* __restrict__ dst,
                                                            int K, int N) {
  __shared__ float tile[32][33];
  size_t mb = (size_t)blockIdx.z * K * N;
  int n0 = blockIdx.x * 32, k0 = blockIdx.y * 32;
  int tx = threadIdx.x & 31, ty = threadIdx.x >> 5;  // 8 rows of 32
#pragma unroll
  for (int i = ty; i < 32; i += 8)
    tile[i][tx] = src[mb + (size_t)(k0 + i) * N + n0 + tx];
  __syncthreads();
#pragma unroll
  for (int i = ty; i < 32; i += 8)
    dst[mb + (size_t)(n0 + i) * K + k0 + tx] = (_Float16)tile[tx][i];
}

// ---------------- RMSNorm (one token per block); writes f16 always, f32 optionally ----------------
__global__ __launch_bounds__(256) void rmsnorm_kernel(const float* __restrict__ x,
                                                      const float* __restrict__ w,
                                                      float* __restrict__ yf,
                                                      _Float16* __restrict__ yh, int H) {
  int t = blockIdx.x;
  const float* xr = x + (size_t)t * H;
  float ss = 0.f;
  for (int i = threadIdx.x; i < H; i += 256) { float v = xr[i]; ss += v * v; }
#pragma unroll
  for (int o = 16; o > 0; o >>= 1) ss += __shfl_xor(ss, o, 32);
  __shared__ float red[8];
  if ((threadIdx.x & 31) == 0) red[threadIdx.x >> 5] = ss;
  __syncthreads();
  if (threadIdx.x < 32) {
    float v = (threadIdx.x < 8) ? red[threadIdx.x] : 0.f;
#pragma unroll
    for (int o = 4; o > 0; o >>= 1) v += __shfl_xor(v, o, 32);
    if (threadIdx.x == 0) red[0] = v;
  }
  __syncthreads();
  float r = rsqrtf(red[0] / (float)H + 1e-6f);
  for (int i = threadIdx.x; i < H; i += 256) {
    float v = xr[i] * r * w[i];
    yh[(size_t)t * H + i] = (_Float16)v;
    if (yf) yf[(size_t)t * H + i] = v;
  }
}

// ---------------- Generic WMMA GEMM: C[M,N] = A[M,K] @ Bt[N,K]^T (+R) ----------------
// Block: 128 thr = 4 waves; block tile 128M x 64N; wave tile 32M x 64N (2 A-frags share each B-frag).
template <bool HASR>
__global__ __launch_bounds__(128) void gemm_f16_kernel(const _Float16* __restrict__ A,
                                                       const _Float16* __restrict__ Bt,
                                                       const float* __restrict__ R,
                                                       float* __restrict__ C,
                                                       int M, int N, int K) {
  int lane = threadIdx.x & 31, wv = threadIdx.x >> 5;
  int hi = lane >> 4, lo = lane & 15;
  int row0 = blockIdx.x * 128 + wv * 32;
  int col0 = blockIdx.y * 64;
  if (row0 >= M) return;  // wave-uniform
  const _Float16* arow0 = A + (size_t)(row0 + lo) * K;
  const _Float16* arow1 = A + (size_t)(row0 + 16 + lo) * K;
  v8f acc0[4] = {}, acc1[4] = {};
  for (int k0 = 0; k0 < K; k0 += 32) {
    v16h a0 = frag_a16(arow0, k0, hi);
    v16h a1 = frag_a16(arow1, k0, hi);
#pragma unroll
    for (int t = 0; t < 4; ++t) {
      v16h b = frag_bt(Bt, K, k0, col0 + t * 16 + lo, hi);
      acc0[t] = wmma32(a0, b, acc0[t]);
      acc1[t] = wmma32(a1, b, acc1[t]);
    }
  }
#pragma unroll
  for (int t = 0; t < 4; ++t) {
    int cb0 = (row0 + hi * 8) * N + col0 + t * 16 + lo;  // all buffers < 2^31 elems
    int cb1 = cb0 + 16 * N;
#pragma unroll
    for (int r = 0; r < 8; ++r) {
      float v0 = acc0[t][r];
      float v1 = acc1[t][r];
      if (HASR) { v0 += R[cb0]; v1 += R[cb1]; }
      C[cb0] = v0;
      C[cb1] = v1;
      cb0 += N;
      cb1 += N;
    }
  }
}

// ---------------- per-head RMSNorm + RoPE: read f32, write f16 ----------------
__global__ __launch_bounds__(128) void qknorm_rope_kernel(const float* __restrict__ x,
                                                          _Float16* __restrict__ xo,
                                                          const float* __restrict__ nw,
                                                          const int* __restrict__ pos,
                                                          int nrows, int heads) {
  int lane = threadIdx.x & 31, wv = threadIdx.x >> 5;
  int row = blockIdx.x * 4 + wv;
  if (row >= nrows) return;
  int s = row / heads;
  const float* p = x + (size_t)row * 64;
  float xlo = p[lane], xhi = p[lane + 32];
  float ss = xlo * xlo + xhi * xhi;
#pragma unroll
  for (int o = 16; o > 0; o >>= 1) ss += __shfl_xor(ss, o, 32);
  float r = rsqrtf(ss * (1.f / 64.f) + 1e-6f);
  xlo *= r * nw[lane];
  xhi *= r * nw[lane + 32];
  float fr = __expf(-(float)lane * 0.03125f * 9.210340371976184f); // 10000^(-lane/32)
  float ang = (float)pos[s] * fr;
  float sn, cs;
  __sincosf(ang, &sn, &cs);
  xo[row * 64 + lane]      = (_Float16)(xlo * cs - xhi * sn);
  xo[row * 64 + lane + 32] = (_Float16)(xhi * cs + xlo * sn);
}

// ---------------- Flash attention step (one 32-key tile) ----------------
// All K and V fragment loads are issued up front (independent registers) so the
// score WMMAs get staggered waits and the V-load latency hides behind the softmax.
template <bool MASKED>
static __device__ __forceinline__ void attn_step(
    int kb, int qb, int hi, int lo, int kvh, int S,
    const _Float16* __restrict__ kh, const _Float16* __restrict__ VT,
    float (*Pw)[33], v16h qa0, v16h qa1, v8f* acc, float& mrow, float& lrow) {
  const int KVH = 4, D = 64;
  // ---- issue all loads first ----
  const _Float16* kr0 = kh + ((size_t)(kb + lo) * KVH + kvh) * D + hi * 16;
  const _Float16* kr1 = kh + ((size_t)(kb + 16 + lo) * KVH + kvh) * D + hi * 16;
  v16h k00 = cat16(*(const v8h*)kr0,        *(const v8h*)(kr0 + 8));
  v16h k01 = cat16(*(const v8h*)(kr0 + 32), *(const v8h*)(kr0 + 40));
  v16h k10 = cat16(*(const v8h*)kr1,        *(const v8h*)(kr1 + 8));
  v16h k11 = cat16(*(const v8h*)(kr1 + 32), *(const v8h*)(kr1 + 40));
  v16h vb[4];
#pragma unroll
  for (int t = 0; t < 4; ++t) {
    const _Float16* vr = VT + (kvh * 64 + t * 16 + lo) * S + kb + hi * 16;
    vb[t] = cat16(*(const v8h*)vr, *(const v8h*)(vr + 8));
  }
  // ---- scores ----
  v8f s0 = {}, s1 = {};
  s0 = wmma32(qa0, k00, s0);
  s0 = wmma32(qa1, k01, s0);
  s1 = wmma32(qa0, k10, s1);
  s1 = wmma32(qa1, k11, s1);
  // scale by D^-0.5; causal mask only in the single diagonal tile (C-layout)
#pragma unroll
  for (int r = 0; r < 8; ++r) {
    if (MASKED) {
      int row = qb + r + hi * 8;
      s0[r] = (kb + lo > row)      ? -1e30f : s0[r] * 0.125f;
      s1[r] = (kb + 16 + lo > row) ? -1e30f : s1[r] * 0.125f;
    } else {
      s0[r] *= 0.125f;
      s1[r] *= 0.125f;
    }
  }
  // C-layout -> transposed via LDS (raw scores)
  LDS_FENCE();
#pragma unroll
  for (int r = 0; r < 8; ++r) {
    Pw[r + hi * 8][lo]      = s0[r];
    Pw[r + hi * 8][lo + 16] = s1[r];
  }
  LDS_FENCE();
  // lane owns q-row `lo`, its half of the 32 columns, in A-fragment element order
  float t0[16];
#pragma unroll
  for (int e = 0; e < 8; ++e) {
    t0[e]     = Pw[lo][e + hi * 16];
    t0[e + 8] = Pw[lo][e + 8 + hi * 16];
  }
  float tm = t0[0];
#pragma unroll
  for (int e = 1; e < 16; ++e) tm = fmaxf(tm, t0[e]);
  tm = fmaxf(tm, __shfl_xor(tm, 16, 32));  // combine the two half-columns of the row
  float mn  = fmaxf(mrow, tm);
  float scl = __expf(mrow - mn);
  mrow = mn;
  float rsum = 0.f;
  v16h pa;
#pragma unroll
  for (int e = 0; e < 16; ++e) {
    float pv = __expf(t0[e] - mn);
    rsum += pv;
    pa[e] = (_Float16)pv;
  }
  rsum += __shfl_xor(rsum, 16, 32);
  lrow = lrow * scl + rsum;
  // broadcast per-row rescale factor back to C-layout via padding column 32
  if (hi == 0) Pw[lo][32] = scl;
  LDS_FENCE();
  float scl8[8];
#pragma unroll
  for (int r = 0; r < 8; ++r) scl8[r] = Pw[r + hi * 8][32];
#pragma unroll
  for (int t = 0; t < 4; ++t)
#pragma unroll
    for (int r = 0; r < 8; ++r) acc[t][r] *= scl8[r];
  // P(16x32) @ V(32x64) using preloaded V fragments
#pragma unroll
  for (int t = 0; t < 4; ++t) acc[t] = wmma32(pa, vb[t], acc[t]);
}

// ---------------- Flash attention: wave = 16 q-rows x full D, online softmax ----------------
__global__ __launch_bounds__(128) void attn_kernel(const _Float16* __restrict__ qh,
                                                   const _Float16* __restrict__ kh,
                                                   const _Float16* __restrict__ VT,
                                                   _Float16* __restrict__ op, int S) {
  const int NH = 16, D = 64;
  int lane = threadIdx.x & 31, wv = threadIdx.x >> 5;
  int hi = lane >> 4, lo = lane & 15;
  int h = blockIdx.x;
  int qb = (blockIdx.y * 4 + wv) * 16;
  if (qb >= S) return;  // wave-uniform
  int kvh = h >> 2;
  __shared__ float P[4][16][33];
  float (*Pw)[33] = P[wv];

  const _Float16* qrow = qh + ((size_t)(qb + lo) * NH + h) * D;
  v16h qa0 = frag_a16(qrow, 0, hi);
  v16h qa1 = frag_a16(qrow, 32, hi);

  v8f acc[4] = {};
  float mrow = -1e30f, lrow = 0.f;

  int kbFull = qb & ~31;  // exactly one masked (diagonal) tile remains
  for (int kb = 0; kb < kbFull; kb += 32)
    attn_step<false>(kb, qb, hi, lo, kvh, S, kh, VT, Pw, qa0, qa1, acc, mrow, lrow);
  attn_step<true>(kbFull, qb, hi, lo, kvh, S, kh, VT, Pw, qa0, qa1, acc, mrow, lrow);

  // bring 1/l back to C-layout rows
  LDS_FENCE();
  if (hi == 0) Pw[lo][32] = 1.0f / lrow;
  LDS_FENCE();
  float linv[8];
#pragma unroll
  for (int r = 0; r < 8; ++r) linv[r] = Pw[r + hi * 8][32];

#pragma unroll
  for (int t = 0; t < 4; ++t)
#pragma unroll
    for (int r = 0; r < 8; ++r)
      op[((qb + r + hi * 8) * NH + h) * D + t * 16 + lo] = (_Float16)(acc[t][r] * linv[r]);
}

// ---------------- Router: logits, sigmoid, grouped top-k ----------------
__global__ __launch_bounds__(128) void router_kernel(const float* __restrict__ x,
                                                     const float* __restrict__ rw,
                                                     const float* __restrict__ rb,
                                                     int* __restrict__ ti,
                                                     float* __restrict__ tw, int H) {
  int t = blockIdx.x;
  __shared__ float part[128];
  __shared__ float lg[16];
  const float* xr = x + (size_t)t * H;
  int e = threadIdx.x >> 3, sub = threadIdx.x & 7;
  float s = 0.f;
  for (int kk = sub; kk < H; kk += 8) s += xr[kk] * rw[(size_t)kk * 16 + e];
  part[threadIdx.x] = s;
  __syncthreads();
  if (sub == 0) {
    float v = 0.f;
    for (int j = 0; j < 8; ++j) v += part[(e << 3) + j];
    lg[e] = v;
  }
  __syncthreads();
  if (threadIdx.x == 0) {
    float sc[16], sf[16];
    for (int i = 0; i < 16; ++i) { sc[i] = 1.f / (1.f + __expf(-lg[i])); sf[i] = sc[i] + rb[i]; }
    float gs[4];
    for (int g = 0; g < 4; ++g) {
      float m1 = -1e30f, m2 = -1e30f;
      for (int j = 0; j < 4; ++j) {
        float v = sf[g * 4 + j];
        if (v > m1) { m2 = m1; m1 = v; } else if (v > m2) m2 = v;
      }
      gs[g] = m1 + m2;
    }
    int g1 = 0; for (int g = 1; g < 4; ++g) if (gs[g] > gs[g1]) g1 = g;
    int g2 = -1;
    for (int g = 0; g < 4; ++g) { if (g == g1) continue; if (g2 < 0 || gs[g] > gs[g2]) g2 = g; }
    bool used[16];
    for (int i = 0; i < 16; ++i) used[i] = !((i >> 2) == g1 || (i >> 2) == g2);
    float wsum = 0.f; int idx[4]; float wv[4];
    for (int j = 0; j < 4; ++j) {
      int bi = 0; float bv = -1e30f;
      for (int i = 0; i < 16; ++i)
        if (!used[i] && sf[i] > bv) { bv = sf[i]; bi = i; }
      used[bi] = true;
      idx[j] = bi; wv[j] = sc[bi]; wsum += sc[bi];
    }
    float inv = 1.f / (wsum + 1e-20f);
    for (int j = 0; j < 4; ++j) { ti[t * 4 + j] = idx[j]; tw[t * 4 + j] = wv[j] * inv; }
  }
}

// ---------------- MoE gather bookkeeping ----------------
__global__ void zero_kernel(int* a, int* b, int n) {
  int i = threadIdx.x;
  if (i < n) { a[i] = 0; b[i] = 0; }
}
__global__ void count_kernel(const int* __restrict__ ti, int* __restrict__ cnt, int n) {
  int i = blockIdx.x * blockDim.x + threadIdx.x;
  if (i < n) atomicAdd(&cnt[ti[i]], 1);
}
__global__ void offsets_kernel(const int* __restrict__ cnt, int* __restrict__ off, int E) {
  if (threadIdx.x == 0) {
    int a = 0;
    for (int e = 0; e < E; ++e) { off[e] = a; a += cnt[e]; }
    off[E] = a;
  }
}
__global__ void fill_kernel(const int* __restrict__ ti, const int* __restrict__ off,
                            int* __restrict__ fil, int* __restrict__ tok,
                            int* __restrict__ slt, int n) {
  int i = blockIdx.x * blockDim.x + threadIdx.x;
  if (i < n) {
    int e = ti[i];
    int s = atomicAdd(&fil[e], 1);
    int p = off[e] + s;
    tok[p] = i >> 2;
    slt[i] = p;
  }
}

// ---------------- Expert up/gate: mid[slot,i] = silu(x@wg_e) * (x@wu_e), f16 out ----------------
__global__ __launch_bounds__(256) void expert_gu_kernel(const _Float16* __restrict__ x,
    const _Float16* __restrict__ wgT, const _Float16* __restrict__ wuT,
    const int* __restrict__ tok, const int* __restrict__ off, const int* __restrict__ cnt,
    _Float16* __restrict__ mid) {
  int e = blockIdx.x;
  int n = cnt[e];
  int tile = blockIdx.y;
  if (tile * 16 >= n) return;  // block-uniform
  int base = off[e] + tile * 16;
  int lane = threadIdx.x & 31, wv = threadIdx.x >> 5;
  int hi = lane >> 4, lo = lane & 15;
  int lim = off[e] + n;
  int slot = base + lo;
  if (slot >= lim) slot = lim - 1;  // pad rows reuse last valid token
  const _Float16* arow = x + (size_t)tok[slot] * 1024;
  const _Float16* Bg = wgT + (size_t)e * 1024 * 512;
  const _Float16* Bu = wuT + (size_t)e * 1024 * 512;
  int col0 = wv * 64;
  v8f ga[4] = {}, ua[4] = {};
  for (int k0 = 0; k0 < 1024; k0 += 32) {
    v16h a = frag_a16(arow, k0, hi);
#pragma unroll
    for (int t = 0; t < 4; ++t) {
      ga[t] = wmma32(a, frag_bt(Bg, 1024, k0, col0 + t * 16 + lo, hi), ga[t]);
      ua[t] = wmma32(a, frag_bt(Bu, 1024, k0, col0 + t * 16 + lo, hi), ua[t]);
    }
  }
#pragma unroll
  for (int t = 0; t < 4; ++t) {
    int col = col0 + t * 16 + lo;
#pragma unroll
    for (int r = 0; r < 8; ++r) {
      int srow = base + r + hi * 8;
      if (srow < lim) {
        float g = ga[t][r], u = ua[t][r];
        mid[srow * 512 + col] = (_Float16)((g / (1.f + __expf(-g))) * u);
      }
    }
  }
}

// ---------------- Expert down: ybuf[slot,h] = mid @ wd_e ; wdT f16 [H][I] ----------------
__global__ __launch_bounds__(256) void expert_down_kernel(const _Float16* __restrict__ mid,
    const _Float16* __restrict__ wdT, const int* __restrict__ off, const int* __restrict__ cnt,
    float* __restrict__ ybuf) {
  int e = blockIdx.x;
  int n = cnt[e];
  int tile = blockIdx.y;
  if (tile * 16 >= n) return;
  int base = off[e] + tile * 16;
  int lane = threadIdx.x & 31, wv = threadIdx.x >> 5;
  int hi = lane >> 4, lo = lane & 15;
  int lim = off[e] + n;
  int ar = base + lo;
  if (ar >= lim) ar = lim - 1;
  const _Float16* arow = mid + (size_t)ar * 512;
  const _Float16* Bd = wdT + (size_t)e * 512 * 1024;
#pragma unroll
  for (int half = 0; half < 2; ++half) {
    int col0 = half * 512 + wv * 64;
    v8f acc[4] = {};
    for (int k0 = 0; k0 < 512; k0 += 32) {
      v16h a = frag_a16(arow, k0, hi);
#pragma unroll
      for (int t = 0; t < 4; ++t)
        acc[t] = wmma32(a, frag_bt(Bd, 512, k0, col0 + t * 16 + lo, hi), acc[t]);
    }
#pragma unroll
    for (int t = 0; t < 4; ++t) {
      int col = col0 + t * 16 + lo;
#pragma unroll
      for (int r = 0; r < 8; ++r) {
        int srow = base + r + hi * 8;
        if (srow < lim) ybuf[srow * 1024 + col] = acc[t][r];
      }
    }
  }
}

// ---------------- shared-expert activation: f16 out ----------------
__global__ void silu_mul_kernel(const float* __restrict__ g, const float* __restrict__ u,
                                _Float16* __restrict__ o, int n) {
  int i = blockIdx.x * blockDim.x + threadIdx.x;
  if (i < n) { float gv = g[i]; o[i] = (_Float16)(gv / (1.f + __expf(-gv)) * u[i]); }
}

// ---------------- final combine (fixed order -> deterministic) ----------------
__global__ __launch_bounds__(256) void combine_kernel(const float* __restrict__ hid,
    const float* __restrict__ sh, const float* __restrict__ ybuf,
    const int* __restrict__ slt, const float* __restrict__ tw,
    float* __restrict__ out, int H) {
  int t = blockIdx.x;
  int s0 = slt[t * 4 + 0], s1 = slt[t * 4 + 1], s2 = slt[t * 4 + 2], s3 = slt[t * 4 + 3];
  float w0 = tw[t * 4 + 0], w1 = tw[t * 4 + 1], w2 = tw[t * 4 + 2], w3 = tw[t * 4 + 3];
  for (int i = threadIdx.x; i < H; i += 256) {
    float v = hid[(size_t)t * H + i] + sh[(size_t)t * H + i];
    v += w0 * ybuf[s0 * H + i];
    v += w1 * ybuf[s1 * H + i];
    v += w2 * ybuf[s2 * H + i];
    v += w3 * ybuf[s3 * H + i];
    out[(size_t)t * H + i] = v;
  }
}

extern "C" void kernel_launch(void* const* d_in, const int* in_sizes, int n_in,
                              void* d_out, int out_size, void* d_ws, size_t ws_size,
                              hipStream_t stream) {
  (void)n_in; (void)out_size; (void)ws_size;
  const float* hs   = (const float*)d_in[0];
  const int*   pos  = (const int*)d_in[1];
  const float* ln1w = (const float*)d_in[2];
  const float* ln2w = (const float*)d_in[3];
  const float* qw   = (const float*)d_in[4];
  const float* kw   = (const float*)d_in[5];
  const float* vw   = (const float*)d_in[6];
  const float* ow   = (const float*)d_in[7];
  const float* qnw  = (const float*)d_in[8];
  const float* knw  = (const float*)d_in[9];
  const float* rw   = (const float*)d_in[10];
  const float* rb   = (const float*)d_in[11];
  const float* wg   = (const float*)d_in[12];
  const float* wu   = (const float*)d_in[13];
  const float* wd   = (const float*)d_in[14];
  const float* sg   = (const float*)d_in[15];
  const float* su   = (const float*)d_in[16];
  const float* sd   = (const float*)d_in[17];
  float* out = (float*)d_out;

  const int H = 1024, NH = 16, KVH = 4, D = 64, E = 16, ISH = 512, I = 512;
  const int S = in_sizes[0] / H;   // 2048
  const int T = S;
  const int NS = T * 4;            // total routed slots

  // ---- workspace carve-up: f32 region, then f16 region, then ints ----
  float* wsf = (float*)d_ws;
  float* qbuf  = wsf; wsf += (size_t)T * NH * D;   // q proj (f32, pre-rope)
  float* kbuf  = wsf; wsf += (size_t)T * KVH * D;  // k proj (f32, pre-rope)
  float* vbuf  = wsf; wsf += (size_t)T * KVH * D;  // v proj (f32)
  float* hid   = wsf; wsf += (size_t)T * H;        // attn residual out
  float* h2f   = wsf; wsf += (size_t)T * H;        // ln2 (f32 for router)
  float* sgb   = wsf; wsf += (size_t)T * ISH;
  float* subf  = wsf; wsf += (size_t)T * ISH;
  float* shout = wsf; wsf += (size_t)T * H;
  float* ybuf  = wsf; wsf += (size_t)NS * H;
  float* tw    = wsf; wsf += (size_t)NS;

  _Float16* wsh = (_Float16*)wsf;
  _Float16* hh    = wsh; wsh += (size_t)T * H;          // ln1 f16
  _Float16* h2h   = wsh; wsh += (size_t)T * H;          // ln2 f16
  _Float16* qh    = wsh; wsh += (size_t)T * NH * D;     // q post-rope f16
  _Float16* kh    = wsh; wsh += (size_t)T * KVH * D;    // k post-rope f16
  _Float16* aouth = wsh; wsh += (size_t)T * NH * D;     // attn out f16
  _Float16* sacth = wsh; wsh += (size_t)T * ISH;        // silu(g)*u f16
  _Float16* midh  = wsh; wsh += (size_t)(NS + 16) * I;  // expert mid f16
  _Float16* qwT = wsh; wsh += (size_t)H * (NH * D);
  _Float16* kwT = wsh; wsh += (size_t)H * (KVH * D);
  _Float16* vwT = wsh; wsh += (size_t)H * (KVH * D);
  _Float16* owT = wsh; wsh += (size_t)(NH * D) * H;
  _Float16* wgT = wsh; wsh += (size_t)E * H * I;
  _Float16* wuT = wsh; wsh += (size_t)E * H * I;
  _Float16* wdT = wsh; wsh += (size_t)E * I * H;
  _Float16* sgT = wsh; wsh += (size_t)H * ISH;
  _Float16* suT = wsh; wsh += (size_t)H * ISH;
  _Float16* sdT = wsh; wsh += (size_t)ISH * H;
  _Float16* VT  = wsh; wsh += (size_t)KVH * D * T;

  int* ip  = (int*)wsh;
  int* ti  = ip; ip += NS;
  int* cnt = ip; ip += E;
  int* off = ip; ip += E + 1;
  int* fil = ip; ip += E;
  int* tok = ip; ip += NS;
  int* slt = ip; ip += NS;

  // ---- weight transpose + f16 convert (Bt[N][K]) ----
  transpose_cvt_kernel<<<dim3((NH * D) / 32, H / 32, 1), 256, 0, stream>>>(qw, qwT, H, NH * D);
  transpose_cvt_kernel<<<dim3((KVH * D) / 32, H / 32, 1), 256, 0, stream>>>(kw, kwT, H, KVH * D);
  transpose_cvt_kernel<<<dim3((KVH * D) / 32, H / 32, 1), 256, 0, stream>>>(vw, vwT, H, KVH * D);
  transpose_cvt_kernel<<<dim3(H / 32, (NH * D) / 32, 1), 256, 0, stream>>>(ow, owT, NH * D, H);
  transpose_cvt_kernel<<<dim3(I / 32, H / 32, E), 256, 0, stream>>>(wg, wgT, H, I);
  transpose_cvt_kernel<<<dim3(I / 32, H / 32, E), 256, 0, stream>>>(wu, wuT, H, I);
  transpose_cvt_kernel<<<dim3(H / 32, I / 32, E), 256, 0, stream>>>(wd, wdT, I, H);
  transpose_cvt_kernel<<<dim3(ISH / 32, H / 32, 1), 256, 0, stream>>>(sg, sgT, H, ISH);
  transpose_cvt_kernel<<<dim3(ISH / 32, H / 32, 1), 256, 0, stream>>>(su, suT, H, ISH);
  transpose_cvt_kernel<<<dim3(H / 32, ISH / 32, 1), 256, 0, stream>>>(sd, sdT, ISH, H);

  // ---- attention block ----
  rmsnorm_kernel<<<T, 256, 0, stream>>>(hs, ln1w, nullptr, hh, H);
  gemm_f16_kernel<false><<<dim3(T / 128, (NH * D) / 64), 128, 0, stream>>>(hh, qwT, nullptr, qbuf, T, NH * D, H);
  gemm_f16_kernel<false><<<dim3(T / 128, (KVH * D) / 64), 128, 0, stream>>>(hh, kwT, nullptr, kbuf, T, KVH * D, H);
  gemm_f16_kernel<false><<<dim3(T / 128, (KVH * D) / 64), 128, 0, stream>>>(hh, vwT, nullptr, vbuf, T, KVH * D, H);
  qknorm_rope_kernel<<<(T * NH) / 4, 128, 0, stream>>>(qbuf, qh, qnw, pos, T * NH, NH);
  qknorm_rope_kernel<<<(T * KVH) / 4, 128, 0, stream>>>(kbuf, kh, knw, pos, T * KVH, KVH);
  transpose_cvt_kernel<<<dim3((KVH * D) / 32, T / 32, 1), 256, 0, stream>>>(vbuf, VT, T, KVH * D);
  attn_kernel<<<dim3(NH, T / 64), 128, 0, stream>>>(qh, kh, VT, aouth, T);
  gemm_f16_kernel<true><<<dim3(T / 128, H / 64), 128, 0, stream>>>(aouth, owT, hs, hid, T, H, NH * D);

  // ---- MoE block ----
  rmsnorm_kernel<<<T, 256, 0, stream>>>(hid, ln2w, h2f, h2h, H);
  router_kernel<<<T, 128, 0, stream>>>(h2f, rw, rb, ti, tw, H);
  zero_kernel<<<1, 64, 0, stream>>>(cnt, fil, E);
  count_kernel<<<(NS + 255) / 256, 256, 0, stream>>>(ti, cnt, NS);
  offsets_kernel<<<1, 1, 0, stream>>>(cnt, off, E);
  fill_kernel<<<(NS + 255) / 256, 256, 0, stream>>>(ti, off, fil, tok, slt, NS);
  expert_gu_kernel<<<dim3(E, T / 16), 256, 0, stream>>>(h2h, wgT, wuT, tok, off, cnt, midh);
  expert_down_kernel<<<dim3(E, T / 16), 256, 0, stream>>>(midh, wdT, off, cnt, ybuf);

  // shared expert
  gemm_f16_kernel<false><<<dim3(T / 128, ISH / 64), 128, 0, stream>>>(h2h, sgT, nullptr, sgb, T, ISH, H);
  gemm_f16_kernel<false><<<dim3(T / 128, ISH / 64), 128, 0, stream>>>(h2h, suT, nullptr, subf, T, ISH, H);
  silu_mul_kernel<<<(T * ISH + 255) / 256, 256, 0, stream>>>(sgb, subf, sacth, T * ISH);
  gemm_f16_kernel<false><<<dim3(T / 128, H / 64), 128, 0, stream>>>(sacth, sdT, nullptr, shout, T, H, ISH);

  combine_kernel<<<T, 256, 0, stream>>>(hid, shout, ybuf, slt, tw, out, H);
}
